// BitLinear_7241314861836
// MI455X (gfx1250) — compile-verified, matrix operand order checked by reading, not run on hardware
//
#include <hip/hip_runtime.h>

// BitLinear: out = (rmsnorm(x) @ ternary(W)^T) * scale
//   x: [4,2048,2048] f32 -> M=8192, K=2048 ; W: [8192,2048] f32 -> N=8192
//   out: [8192, 8192] f32
//
// MI455X plan: ternary weights exact in f16, x_norm -> f16, GEMM on
// V_WMMA_F32_16X16X32_F16. f16 A+B (64 MiB) is L2-resident; HBM traffic is
// dominated by the 256 MiB output. GEMM inner loop feeds LDS with
// GLOBAL_LOAD_ASYNC_TO_LDS_B128 (ASYNCcnt DMA) double-buffered so the DMA for
// tile k+1 overlaps the WMMAs on tile k.
//
// Workspace (~64 MiB):
//   [0,32Mi)      Xq f16 [8192][2048]
//   [32Mi,64Mi)   Wq f16 [8192][2048]
//   [64Mi,+8Ki)   2048 f32 partials;  then {scale, 1/scale}

typedef __attribute__((ext_vector_type(16))) _Float16 v16h;
typedef __attribute__((ext_vector_type(8)))  _Float16 v8h;
typedef __attribute__((ext_vector_type(8)))  float    v8f;
typedef __attribute__((ext_vector_type(4)))  float    v4f;

#define M_TOT 8192
#define K_DIM 2048
#define N_TOT 8192
#define NK    (N_TOT * K_DIM)

// ---------------------------------------------------------------- reductions
__global__ __launch_bounds__(256)
void reduce_abs_kernel(const float* __restrict__ w, float* __restrict__ partials) {
  __shared__ float red[256];
  const int tid = threadIdx.x;
  float s = 0.0f;
  const size_t base = (size_t)blockIdx.x * 8192;
#pragma unroll
  for (int j = 0; j < 8; ++j) {
    v4f v = *(const v4f*)(w + base + (size_t)j * 1024 + (size_t)tid * 4);
    s += fabsf(v[0]) + fabsf(v[1]) + fabsf(v[2]) + fabsf(v[3]);
  }
  red[tid] = s;
  __syncthreads();
  for (int o = 128; o > 0; o >>= 1) {
    if (tid < o) red[tid] += red[tid + o];
    __syncthreads();
  }
  if (tid == 0) partials[blockIdx.x] = red[0];
}

__global__ __launch_bounds__(256)
void finalize_scale_kernel(const float* __restrict__ partials, float* __restrict__ scale) {
  __shared__ float red[256];
  const int tid = threadIdx.x;
  float s = 0.0f;
#pragma unroll
  for (int j = 0; j < 8; ++j) s += partials[tid + j * 256];
  red[tid] = s;
  __syncthreads();
  for (int o = 128; o > 0; o >>= 1) {
    if (tid < o) red[tid] += red[tid + o];
    __syncthreads();
  }
  if (tid == 0) {
    float sc = fmaxf(red[0] * (1.0f / (float)NK), 1e-5f);
    scale[0] = sc;
    scale[1] = 1.0f / sc;
  }
}

// ------------------------------------------------------- weight quantization
__global__ __launch_bounds__(256)
void quantize_w_kernel(const float* __restrict__ w, const float* __restrict__ scale,
                       _Float16* __restrict__ wq) {
  const float inv = scale[1];
  const size_t i0 = ((size_t)blockIdx.x * 256 + threadIdx.x) * 8;
  v4f w0 = *(const v4f*)(w + i0);
  v4f w1 = *(const v4f*)(w + i0 + 4);
  v8h q;
#pragma unroll
  for (int j = 0; j < 4; ++j) {
    q[j]     = (_Float16)rintf(fminf(fmaxf(w0[j] * inv, -1.0f), 1.0f));
    q[j + 4] = (_Float16)rintf(fminf(fmaxf(w1[j] * inv, -1.0f), 1.0f));
  }
  *(v8h*)(wq + i0) = q;
}

// ------------------------------------------------------------------- rmsnorm
__global__ __launch_bounds__(256)
void rmsnorm_f16_kernel(const float* __restrict__ x, const float* __restrict__ gamma,
                        _Float16* __restrict__ xq) {
  __shared__ float red[256];
  const int tid = threadIdx.x;
  const size_t row = blockIdx.x;
  const float* xr = x + row * (size_t)K_DIM;
  float v[8];
  float s = 0.0f;
#pragma unroll
  for (int j = 0; j < 8; ++j) {
    v[j] = xr[tid + j * 256];
    s += v[j] * v[j];
  }
  red[tid] = s;
  __syncthreads();
  for (int o = 128; o > 0; o >>= 1) {
    if (tid < o) red[tid] += red[tid + o];
    __syncthreads();
  }
  const float inv = rsqrtf(red[0] * (1.0f / (float)K_DIM) + 1e-5f);
  _Float16* xo = xq + row * (size_t)K_DIM;
#pragma unroll
  for (int j = 0; j < 8; ++j) {
    const int c = tid + j * 256;
    xo[c] = (_Float16)(v[j] * inv * gamma[c]);
  }
}

// ---------------------------------------------------------------- WMMA GEMM
// Block tile 128(M) x 256(N), BK=64 (two WMMA K-steps per LDS tile).
// 256 threads = 8 waves; wave grid 2(M) x 4(N); wave tile 64x64 = 4x4 frags
// -> 32 WMMAs between barriers, 1 ds_load_b128 per WMMA.
// LDS rows padded 64 -> 72 f16 (144 B = 36 dwords): 16-lane b128 fragment
// reads start at banks 4*(9i mod 16) = all multiples of 4 -> conflict-free.
// Double-buffered; tiles filled by GLOBAL_LOAD_ASYNC_TO_LDS_B128 DMA.
#define BM 128
#define BN 256
#define BK 64
#define LDSW 72
#define A_ELEMS (BM * LDSW)            // 9216 f16
#define BUF_ELEMS ((BM + BN) * LDSW)   // 27648 f16 per buffer

__device__ __forceinline__ void async_b128(unsigned lds_addr, unsigned gbyte_off,
                                           const _Float16* gbase) {
  // GVS form: SGPR 64-bit base + 32-bit VGPR byte offset; LDS dest addr in VGPR.
  asm volatile("global_load_async_to_lds_b128 %0, %1, %2"
               :
               : "v"(lds_addr), "v"(gbyte_off), "s"(gbase)
               : "memory");
}

__global__ __launch_bounds__(256)
void gemm_wmma_kernel(const _Float16* __restrict__ Xq, const _Float16* __restrict__ Wq,
                      const float* __restrict__ scale, float* __restrict__ out) {
  __shared__ _Float16 smem[2 * BUF_ELEMS];   // 110,592 B

  const int tid   = threadIdx.x;
  const int lane  = tid & 31;
  const int wv    = tid >> 5;       // 0..7
  const int waveM = wv & 1;         // 2 waves along M, 64 rows each
  const int waveN = wv >> 1;        // 4 waves along N, 64 cols each
  const int half  = lane >> 4;      // f16 A/B fragment: lane-half K phase
  const int lr    = lane & 15;

  const int bm = blockIdx.y * BM;
  const int bn = blockIdx.x * BN;

  const _Float16* gA = Xq + (size_t)bm * K_DIM;
  const _Float16* gB = Wq + (size_t)bn * K_DIM;

  // DMA mapping: 8 lanes cover one 128 B row-chunk of BK=64 f16
  const int r0   = tid >> 3;               // 0..31
  const unsigned coff = (tid & 7) * 8;     // f16 element offset in row

  const unsigned lds_base = (unsigned)(size_t)(&smem[0]);

  v8f acc[4][4];
#pragma unroll
  for (int fm = 0; fm < 4; ++fm)
#pragma unroll
    for (int fn = 0; fn < 4; ++fn)
      acc[fm][fn] = (v8f){0.f, 0.f, 0.f, 0.f, 0.f, 0.f, 0.f, 0.f};

  // ---- issue DMA for one (buffer, k0) tile: 12 x b128 per thread ----------
  auto issue_tile = [&](int buf, int k0) {
    const unsigned ldsA = lds_base + (unsigned)(buf * BUF_ELEMS * 2);
    const unsigned ldsB = ldsA + (unsigned)(A_ELEMS * 2);
#pragma unroll
    for (int p = 0; p < 4; ++p) {          // A: 128 rows
      const int row = r0 + p * 32;
      async_b128(ldsA + (unsigned)((row * LDSW + coff) * 2),
                 (unsigned)((row * K_DIM + k0 + coff) * 2), gA);
    }
#pragma unroll
    for (int p = 0; p < 8; ++p) {          // B: 256 rows
      const int row = r0 + p * 32;
      async_b128(ldsB + (unsigned)((row * LDSW + coff) * 2),
                 (unsigned)((row * K_DIM + k0 + coff) * 2), gB);
    }
  };

  const int niter = K_DIM / BK;            // 32
  issue_tile(0, 0);

  for (int it = 0; it < niter; ++it) {
    if (it + 1 < niter) {
      issue_tile((it + 1) & 1, (it + 1) * BK);
      // 24 outstanding; oldest 12 (current buffer) must complete
      asm volatile("s_wait_asynccnt 0xc" ::: "memory");
    } else {
      asm volatile("s_wait_asynccnt 0x0" ::: "memory");
    }
    __syncthreads();                       // all waves' DMA for this buffer done

    const _Float16* bufA = &smem[(it & 1) * BUF_ELEMS];
    const _Float16* bufB = bufA + A_ELEMS;

#pragma unroll
    for (int ks = 0; ks < BK; ks += 32) {
      // ISA 7.12.2 f16 16x32 A/B layout: lanes 0-15 hold K[c..c+7],K[c+16..23];
      // lanes 16-31 hold K[c+8..15],K[c+24..31]. A (MxK) and B (NxK = W^T) use
      // identical per-lane addressing since both are K-contiguous.
      const int c0 = ks + half * 8;
      v16h a[4], b[4];
#pragma unroll
      for (int fm = 0; fm < 4; ++fm) {
        const _Float16* p = &bufA[(waveM * 64 + fm * 16 + lr) * LDSW + c0];
        v8h lo = *(const v8h*)p;
        v8h hi = *(const v8h*)(p + 16);
        a[fm] = __builtin_shufflevector(lo, hi, 0, 1, 2, 3, 4, 5, 6, 7,
                                        8, 9, 10, 11, 12, 13, 14, 15);
      }
#pragma unroll
      for (int fn = 0; fn < 4; ++fn) {
        const _Float16* p = &bufB[(waveN * 64 + fn * 16 + lr) * LDSW + c0];
        v8h lo = *(const v8h*)p;
        v8h hi = *(const v8h*)(p + 16);
        b[fn] = __builtin_shufflevector(lo, hi, 0, 1, 2, 3, 4, 5, 6, 7,
                                        8, 9, 10, 11, 12, 13, 14, 15);
      }
#pragma unroll
      for (int fm = 0; fm < 4; ++fm)
#pragma unroll
        for (int fn = 0; fn < 4; ++fn)
          acc[fm][fn] = __builtin_amdgcn_wmma_f32_16x16x32_f16(
              false, a[fm], false, b[fn], (short)0, acc[fm][fn], false, false);
    }
    __syncthreads();   // all waves done reading before buffer is DMA-overwritten
  }

  // epilogue: C layout (32-bit 16x16): VGPR r -> M = r + 8*half, N = lr
  const float s = scale[0];
#pragma unroll
  for (int fm = 0; fm < 4; ++fm)
#pragma unroll
    for (int fn = 0; fn < 4; ++fn)
#pragma unroll
      for (int r = 0; r < 8; ++r) {
        const int row = bm + waveM * 64 + fm * 16 + half * 8 + r;
        const int col = bn + waveN * 64 + fn * 16 + lr;
        out[(size_t)row * N_TOT + col] = acc[fm][fn][r] * s;
      }
}

// ------------------------------------------------------------------- driver
extern "C" void kernel_launch(void* const* d_in, const int* in_sizes, int n_in,
                              void* d_out, int out_size, void* d_ws, size_t ws_size,
                              hipStream_t stream) {
  const float* x     = (const float*)d_in[0];   // [4,2048,2048]
  const float* w     = (const float*)d_in[1];   // [8192,2048]
  const float* gamma = (const float*)d_in[2];   // [2048]
  float* out = (float*)d_out;

  char* ws = (char*)d_ws;
  const size_t XQ_BYTES = (size_t)M_TOT * K_DIM * sizeof(_Float16);  // 32 MiB
  const size_t WQ_BYTES = (size_t)N_TOT * K_DIM * sizeof(_Float16);  // 32 MiB
  _Float16* Xq = (_Float16*)ws;
  _Float16* Wq = (_Float16*)(ws + XQ_BYTES);
  float* partials = (float*)(ws + XQ_BYTES + WQ_BYTES);
  float* scale    = partials + 2048;

  reduce_abs_kernel<<<2048, 256, 0, stream>>>(w, partials);
  finalize_scale_kernel<<<1, 256, 0, stream>>>(partials, scale);
  quantize_w_kernel<<<NK / (256 * 8), 256, 0, stream>>>(w, scale, Wq);
  rmsnorm_f16_kernel<<<M_TOT, 256, 0, stream>>>(x, gamma, Xq);
  gemm_wmma_kernel<<<dim3(N_TOT / BN, M_TOT / BM), 256, 0, stream>>>(Xq, Wq, scale, out);
}
